// SpatioTemporalAttention_68152541053302
// MI455X (gfx1250) — matrix operand order repaired and checked by reading, not compiled, not run on hardware
//
#include <hip/hip_runtime.h>
#include <hip/hip_bf16.h>

// ---------------------------------------------------------------------------
// Types for CDNA5 WMMA (wave32): v16 bf16 A/B fragments, v8 f32 accumulator.
// ---------------------------------------------------------------------------
typedef __bf16 bf16_t;
typedef bf16_t bf16x8  __attribute__((ext_vector_type(8)));
typedef bf16_t bf16x16 __attribute__((ext_vector_type(16)));
typedef float  f32x8   __attribute__((ext_vector_type(8)));
typedef unsigned int u32x4 __attribute__((ext_vector_type(4)));
typedef int          i32x4 __attribute__((ext_vector_type(4)));
typedef int          i32x8 __attribute__((ext_vector_type(8)));

#define DEV static __device__ __forceinline__

DEV unsigned short f32_to_bf16_raw(float f) {
  unsigned int u = __float_as_uint(f);
  u += 0x7FFFu + ((u >> 16) & 1u);        // round-to-nearest-even
  return (unsigned short)(u >> 16);
}
DEV float bf16_raw_to_f32(unsigned short h) {
  return __uint_as_float(((unsigned int)h) << 16);
}

DEV bf16x16 make_frag(const unsigned short* p0, const unsigned short* p1) {
  bf16x8 a = *(const bf16x8*)p0;
  bf16x8 b = *(const bf16x8*)p1;
  return __builtin_shufflevector(a, b, 0,1,2,3,4,5,6,7,8,9,10,11,12,13,14,15);
}

// A fragment 16x32 (MxK) bf16, row-major source with leading dim `ld`.
// lane = (lo,hi): row m; elems i<8 -> K = k + hi*8 + i ; i>=8 -> K = k+16+hi*8+(i-8)
DEV bf16x16 load_a(const unsigned short* base, int ld, int m, int k, int hi) {
  const unsigned short* r = base + (size_t)m * ld + k + hi * 8;
  return make_frag(r, r + 16);
}
// B fragment 32x16 (KxN) bf16 from an [N][K] row-major source (leading dim `ld`).
// lane = (lo,hi): column n; elems i<8 -> K = k + hi*8 + i ; i>=8 -> K = k+16+hi*8+(i-8)
DEV bf16x16 load_b(const unsigned short* base, int ld, int n, int k, int hi) {
  const unsigned short* r = base + (size_t)n * ld + k + hi * 8;
  return make_frag(r, r + 16);
}

DEV f32x8 wmma_bf16(bf16x16 a, bf16x16 b, f32x8 c) {
  return __builtin_amdgcn_wmma_f32_16x16x32_bf16(false, a, false, b,
                                                 (short)0, c, false, false);
}

// ---------------------------------------------------------------------------
// Tensor Data Mover: DMA a 2-D bf16 tile [rows x cols] (row stride
// `row_stride_elems` elements in global) into LDS, packed row-major
// [rows][cols]. Descriptor per CDNA5 ISA.
// Tracked with TENSORcnt (in-order per wave). 6-arg builtin on this toolchain:
//   (u32x4 g0, i32x8 g1, i32x4 g2, i32x4 g3, i32x8 g4, i32 cpol)
// groups 2-4 unused for a 2-D tile -> zero-filled.
// ---------------------------------------------------------------------------
DEV void tdm_load_2d(const unsigned short* lds_dst, const unsigned short* gsrc,
                     int rows, int cols, int row_stride_elems) {
  unsigned long long ga = (unsigned long long)(uintptr_t)gsrc;
  unsigned lds_off = (unsigned)(uintptr_t)(const void*)lds_dst; // LDS = addr[31:0]
  u32x4 g0;
  g0[0] = 1u;                                           // count=1, user descriptor
  g0[1] = lds_off;                                      // lds_addr (bytes)
  g0[2] = (unsigned)ga;                                 // global_addr[31:0]
  g0[3] = (unsigned)((ga >> 32) & 0x01FFFFFFull)        // global_addr[56:32]
        | (2u << 30);                                   // type = 2 ("image")
  const unsigned td1 = 1u << 20;                        // tensor_dim1 (OOB never hit)
  unsigned str = (unsigned)row_stride_elems;
  i32x8 g1;
  g1[0] = (int)0x00010000u;                             // data_size = 2 bytes
  g1[1] = (int)((str & 0xFFFFu) << 16);                 // tensor_dim0[15:0] @bits63:48
  g1[2] = (int)(((str >> 16) & 0xFFFFu)                 // tensor_dim0[31:16]
        | ((td1 & 0xFFFFu) << 16));                     // tensor_dim1[15:0]
  g1[3] = (int)(((td1 >> 16) & 0xFFFFu)                 // tensor_dim1[31:16]
        | (((unsigned)cols & 0xFFFFu) << 16));          // tile_dim0 = cols
  g1[4] = rows & 0xFFFF;                                // tile_dim1 (tile_dim2=0 -> 2D)
  g1[5] = (int)str;                                     // tensor_dim0_stride[31:0]
  g1[6] = 0;                                            // stride hi / dim1_stride
  g1[7] = 0;
  i32x4 gz4 = {0, 0, 0, 0};
  i32x8 gz8 = {0, 0, 0, 0, 0, 0, 0, 0};
  __builtin_amdgcn_tensor_load_to_lds(g0, g1, gz4, gz4, gz8, 0);
}

// ---------------------------------------------------------------------------
// Problem constants
// ---------------------------------------------------------------------------
#define BF_FRAMES 128     // B*T
#define S_TOK     256     // H*W tokens per frame
#define D_MODEL_C 1152
#define E_QKV     1920    // 12*96 + 2*4*96
#define N_HEADS   12
#define HEAD_D    96
#define M_ROWS    32768   // BF_FRAMES * S_TOK
#define KC        128     // GEMM K-chunk staged per TDM transfer
#define A_TILE    (128 * KC)   // elements per A buffer
#define B_TILE    (64 * KC)    // elements per B buffer

// ---------------------------------------------------------------------------
// fp32 -> bf16 conversion (4 elements/thread)
// ---------------------------------------------------------------------------
__global__ void k_f32_to_bf16(const float* __restrict__ src,
                              unsigned short* __restrict__ dst, int n) {
  int i = (blockIdx.x * blockDim.x + threadIdx.x) * 4;
  if (i + 3 < n) {
    float4 v = *(const float4*)(src + i);
    dst[i + 0] = f32_to_bf16_raw(v.x);
    dst[i + 1] = f32_to_bf16_raw(v.y);
    dst[i + 2] = f32_to_bf16_raw(v.z);
    dst[i + 3] = f32_to_bf16_raw(v.w);
  }
}

// ---------------------------------------------------------------------------
// bf16 GEMM: C[M,N] = A[M,K] * B[N,K]^T  (fp32 accumulate), TDM-fed.
// block = 256 threads = 8 waves; block tile 128(M) x 64(N); wave tile 32x32.
// Double-buffered LDS staging: A-tile 128xKC + B-tile 64xKC per chunk,
// DMA'd by wave 0 via tensor_load_to_lds, synchronized with s_wait_tensorcnt.
// Dynamic LDS = (A_TILE + B_TILE) * 2 bytes * 2 buffers = 96 KB.
// LDS map: [ A0 | A1 | B0 | B1 ]  (offset arithmetic only -- no pointer
// arrays; clang folds those into addrspacecast static initializers that
// ld.lld rejects).
// ---------------------------------------------------------------------------
__global__ void k_gemm_bf16(const unsigned short* __restrict__ A,
                            const unsigned short* __restrict__ B,
                            void* __restrict__ C,
                            int M, int N, int K, int out_f32) {
  extern __shared__ unsigned short smem_u16[];

  const int lane = threadIdx.x & 31;
  const int wid  = threadIdx.x >> 5;
  const int lo = lane & 15, hi = lane >> 4;
  const int mb = blockIdx.y * 128;
  const int nb = blockIdx.x * 64;
  const int wm = (wid >> 1) * 32;            // wave M offset in block tile
  const int wn = (wid & 1) * 32;             // wave N offset in block tile
  const int nch = K / KC;

  if (wid == 0) {
    tdm_load_2d(smem_u16,              A + (size_t)mb * K, 128, KC, K);
    tdm_load_2d(smem_u16 + 2 * A_TILE, B + (size_t)nb * K, 64,  KC, K);
  }

  f32x8 acc[2][2] = {};
  for (int c = 0; c < nch; ++c) {
    const int cur = c & 1, nxt = cur ^ 1;
    if (wid == 0) {
      if (c + 1 < nch) {
        tdm_load_2d(smem_u16 + nxt * A_TILE,
                    A + (size_t)mb * K + (c + 1) * KC, 128, KC, K);
        tdm_load_2d(smem_u16 + 2 * A_TILE + nxt * B_TILE,
                    B + (size_t)nb * K + (c + 1) * KC, 64,  KC, K);
        __builtin_amdgcn_s_wait_tensorcnt(2);   // chunk c done (in-order)
      } else {
        __builtin_amdgcn_s_wait_tensorcnt(0);
      }
    }
    __syncthreads();                            // chunk c visible to all waves
    const unsigned short* tA = smem_u16 + cur * A_TILE;
    const unsigned short* tB = smem_u16 + 2 * A_TILE + cur * B_TILE;
    for (int kk = 0; kk < KC; kk += 32) {
      bf16x16 a0 = load_a(tA, KC, wm + lo,      kk, hi);
      bf16x16 a1 = load_a(tA, KC, wm + 16 + lo, kk, hi);
      bf16x16 b0 = load_b(tB, KC, wn + lo,      kk, hi);
      bf16x16 b1 = load_b(tB, KC, wn + 16 + lo, kk, hi);
      acc[0][0] = wmma_bf16(a0, b0, acc[0][0]);
      acc[0][1] = wmma_bf16(a0, b1, acc[0][1]);
      acc[1][0] = wmma_bf16(a1, b0, acc[1][0]);
      acc[1][1] = wmma_bf16(a1, b1, acc[1][1]);
    }
    __syncthreads();                            // done reading before overwrite
  }

  for (int mi = 0; mi < 2; ++mi)
    for (int ni = 0; ni < 2; ++ni)
      for (int j = 0; j < 8; ++j) {
        int row = mb + wm + mi * 16 + hi * 8 + j;  // C layout: M = hi*8 + j
        int col = nb + wn + ni * 16 + lo;          //           N = lo
        float v = acc[mi][ni][j];
        if (out_f32) ((float*)C)[(size_t)row * N + col] = v;
        else ((unsigned short*)C)[(size_t)row * N + col] = f32_to_bf16_raw(v);
      }
}

// ---------------------------------------------------------------------------
// Fused qk-norm + 2D RoPE + GQA re-layout.
// One wave per 96-dim head vector (lane holds dims {l, l+32, l+64}).
//   q -> qb [BF][12][S][96]   (normed + roped)
//   k -> kb [BF][4 ][S][96]   (normed + roped)
//   v -> vb [BF][4 ][96][S]   (raw, pre-transposed for PV WMMA B-frags)
// ---------------------------------------------------------------------------
__global__ void k_normrope(const unsigned short* __restrict__ qkv,
                           unsigned short* __restrict__ qb,
                           unsigned short* __restrict__ kb,
                           unsigned short* __restrict__ vb) {
  const int lane = threadIdx.x & 31;
  const int wid  = threadIdx.x >> 5;
  int vid  = blockIdx.x * 8 + wid;           // [0, M_ROWS*20)
  int row  = vid / 20;
  int slot = vid - row * 20;                 // 0..11 q, 12..15 k, 16..19 v
  if (row >= M_ROWS) return;
  int bf = row >> 8, s = row & 255;

  const unsigned short* src;
  if (slot < 12)       src = qkv + (size_t)row * E_QKV + slot * 96;
  else if (slot < 16)  src = qkv + (size_t)row * E_QKV + 1152 + (slot - 12) * 96;
  else                 src = qkv + (size_t)row * E_QKV + 1536 + (slot - 16) * 96;

  unsigned short r0 = src[lane], r1 = src[lane + 32], r2 = src[lane + 64];

  if (slot >= 16) {                          // v: transpose copy only
    int g = slot - 16;
    unsigned short* dst = vb + (((size_t)bf * 4 + g) * 96) * 256;
    dst[(size_t)(lane)      * 256 + s] = r0;
    dst[(size_t)(lane + 32) * 256 + s] = r1;
    dst[(size_t)(lane + 64) * 256 + s] = r2;
    return;
  }

  float v0 = bf16_raw_to_f32(r0);
  float v1 = bf16_raw_to_f32(r1);
  float v2 = bf16_raw_to_f32(r2);
  float ss = v0 * v0 + v1 * v1 + v2 * v2;
  for (int m = 16; m >= 1; m >>= 1) ss += __shfl_xor(ss, m, 32);
  float sc = 1.0f / fmaxf(sqrtf(ss), 1e-10f);
  v0 *= sc; v1 *= sc; v2 *= sc;

  // 2D RoPE on dims [0,32): out[j] = x1[j]c - x2[j]s ; out[16+j] = x1[j]s + x2[j]c
  int   j    = lane & 15;
  float x1   = __shfl(v0, 2 * j, 32);        // dim 2j
  float x2   = __shfl(v0, 2 * j + 1, 32);    // dim 2j+1
  float invf = expf(-(float)j * (9.210340371976184f / 16.0f)); // 10000^(-j/16)
  float pos  = (float)((s >> 4) + (s & 15)); // h_pos + w_pos
  float ang  = pos * invf;
  float c = cosf(ang), sn = sinf(ang);
  float rv = (lane < 16) ? (x1 * c - x2 * sn) : (x1 * sn + x2 * c);

  unsigned short o0 = f32_to_bf16_raw(rv);
  unsigned short o1 = f32_to_bf16_raw(v1);
  unsigned short o2 = f32_to_bf16_raw(v2);
  unsigned short* dst;
  if (slot < 12) dst = qb + (((size_t)bf * 12 + slot)        * 256 + s) * 96;
  else           dst = kb + (((size_t)bf * 4  + (slot - 12)) * 256 + s) * 96;
  dst[lane] = o0; dst[lane + 32] = o1; dst[lane + 64] = o2;
}

// ---------------------------------------------------------------------------
// Attention: one block (256 thr / 8 waves) per (frame, head); wave owns 32 q-rows.
// Per-wave 32x256 fp32 score slice in LDS (8 waves -> 256 KB dynamic LDS),
// fp32 softmax compacted in place to bf16 (unnormalized exp), PV via WMMA,
// final 1/rowsum applied to the output fragments.
// ---------------------------------------------------------------------------
__global__ void k_attention(const unsigned short* __restrict__ q,
                            const unsigned short* __restrict__ kx,
                            const unsigned short* __restrict__ vx,
                            unsigned short* __restrict__ ob) {
  extern __shared__ char smem_raw[];
  float* s_f = (float*)smem_raw;             // 8 * 32 * 256 fp32 = 256 KB
  const int lane = threadIdx.x & 31;
  const int w    = threadIdx.x >> 5;
  const int lo = lane & 15, hi = lane >> 4;
  const int bf = blockIdx.x / N_HEADS;
  const int h  = blockIdx.x % N_HEADS;
  const int g  = h / 3;                      // heads-per-group = 3

  const unsigned short* qh = q  + (((size_t)bf * 12 + h) * 256) * 96;
  const unsigned short* kg = kx + (((size_t)bf * 4  + g) * 256) * 96;
  const unsigned short* vg = vx + (((size_t)bf * 4  + g) * 96)  * 256; // [96][256]
  __builtin_prefetch(qh, 0, 3);              // global_prefetch_b8
  __builtin_prefetch(kg, 0, 3);
  __builtin_prefetch(vg, 0, 3);
  float* slice = s_f + (size_t)w * 32 * 256;
  const float scale = 0.10206207261596575f;  // 96^-0.5

  // Phase 1: scores[w*32 .. w*32+31][0..255] = scale * q @ k^T
  for (int nt = 0; nt < 16; ++nt) {
    f32x8 acc0 = {}, acc1 = {};
    for (int d = 0; d < 96; d += 32) {
      bf16x16 a0 = load_a(qh, 96, w * 32 + lo,      d, hi);
      bf16x16 a1 = load_a(qh, 96, w * 32 + 16 + lo, d, hi);
      bf16x16 b  = load_b(kg, 96, nt * 16 + lo,     d, hi);
      acc0 = wmma_bf16(a0, b, acc0);
      acc1 = wmma_bf16(a1, b, acc1);
    }
    for (int jj = 0; jj < 8; ++jj) {
      slice[(hi * 8 + jj)      * 256 + nt * 16 + lo] = acc0[jj] * scale;
      slice[(16 + hi * 8 + jj) * 256 + nt * 16 + lo] = acc1[jj] * scale;
    }
  }
  __syncthreads();

  // Phase 2: per-lane row softmax; compact fp32 -> bf16 in place.
  // Row r fp32 occupies bytes [r*1024, r*1024+1024); bf16 row rewrites bytes
  // [r*1024, r*1024+512) -- write offset 2c always trails read offset 4c.
  float* rowp = slice + (size_t)lane * 256;
  float mx = -3.0e38f;
  for (int c = 0; c < 256; ++c) mx = fmaxf(mx, rowp[c]);
  unsigned short* rowb = (unsigned short*)rowp;
  float sum = 0.0f;
  for (int c = 0; c < 256; ++c) {
    float e = __expf(rowp[c] - mx);
    sum += e;
    rowb[c] = f32_to_bf16_raw(e);
  }
  float rcp = 1.0f / sum;
  __syncthreads();

  // Phase 3: o = (exp_scores @ v) * (1/sum). attn rows pitch = 512 ushorts.
  unsigned short* sb = (unsigned short*)slice;
  for (int nt = 0; nt < 6; ++nt) {
    f32x8 acc0 = {}, acc1 = {};
    for (int kk = 0; kk < 256; kk += 32) {
      bf16x16 a0 = load_a(sb, 512, lo,           kk, hi);
      bf16x16 a1 = load_a(sb, 512, 16 + lo,      kk, hi);
      bf16x16 b  = load_b(vg, 256, nt * 16 + lo, kk, hi);
      acc0 = wmma_bf16(a0, b, acc0);
      acc1 = wmma_bf16(a1, b, acc1);
    }
    for (int jj = 0; jj < 8; ++jj) {
      int r0 = hi * 8 + jj, r1 = 16 + hi * 8 + jj;
      float o0 = acc0[jj] * __shfl(rcp, r0, 32);
      float o1 = acc1[jj] * __shfl(rcp, r1, 32);
      int t0 = w * 32 + r0, t1 = w * 32 + r1;
      int col = h * 96 + nt * 16 + lo;
      ob[((size_t)bf * 256 + t0) * 1152 + col] = f32_to_bf16_raw(o0);
      ob[((size_t)bf * 256 + t1) * 1152 + col] = f32_to_bf16_raw(o1);
    }
  }
}

// ---------------------------------------------------------------------------
// Host orchestration
// ---------------------------------------------------------------------------
extern "C" void kernel_launch(void* const* d_in, const int* in_sizes, int n_in,
                              void* d_out, int out_size, void* d_ws, size_t ws_size,
                              hipStream_t stream) {
  (void)in_sizes; (void)n_in; (void)out_size; (void)ws_size;
  const float* x     = (const float*)d_in[0];
  const float* w_qkv = (const float*)d_in[1];
  const float* w_o   = (const float*)d_in[2];
  // d_in[3]/d_in[4] = H_patch/W_patch = 16, hardcoded.

  constexpr size_t M = M_ROWS, D = D_MODEL_C, E = E_QKV;
  char* ws = (char*)d_ws;
  size_t off = 0;
  auto carve = [&](size_t bytes) -> char* {
    char* p = ws + off;
    off += (bytes + 255) & ~(size_t)255;
    return p;
  };
  unsigned short* xb   = (unsigned short*)carve(M * D * 2);        // 75.5 MB
  unsigned short* wqb  = (unsigned short*)carve(E * D * 2);        //  4.4 MB
  unsigned short* wob  = (unsigned short*)carve(D * D * 2);        //  2.7 MB
  unsigned short* qkvb = (unsigned short*)carve(M * E * 2);        // 125.8 MB (reused as o)
  unsigned short* qb   = (unsigned short*)carve(M * 1152 * 2);     // 75.5 MB
  unsigned short* kb   = (unsigned short*)carve(M * 384 * 2);      // 25.2 MB
  unsigned short* vb   = (unsigned short*)carve(M * 384 * 2);      // 25.2 MB

  const unsigned gemm_lds = (A_TILE + B_TILE) * 2 * 2;             // 96 KB

  // 1) precision lowering to bf16
  k_f32_to_bf16<<<(unsigned)(M * D / 1024), 256, 0, stream>>>(x,     xb,  (int)(M * D));
  k_f32_to_bf16<<<(unsigned)(E * D / 1024), 256, 0, stream>>>(w_qkv, wqb, (int)(E * D));
  k_f32_to_bf16<<<(unsigned)(D * D / 1024), 256, 0, stream>>>(w_o,   wob, (int)(D * D));

  // 2) qkv = x @ w_qkv^T   (M x E, K = D), TDM-fed WMMA GEMM
  k_gemm_bf16<<<dim3((unsigned)(E / 64), (unsigned)(M / 128)), 256, gemm_lds, stream>>>(
      xb, wqb, qkvb, (int)M, (int)E, (int)D, 0);

  // 3) qk-norm + RoPE + GQA re-layout (one wave per head vector)
  k_normrope<<<(unsigned)(M * 20 / 8), 256, 0, stream>>>(qkvb, qb, kb, vb);

  // 4) attention; o overwrites the (consumed) qkv buffer
  unsigned short* o_buf = qkvb;
  k_attention<<<BF_FRAMES * N_HEADS, 256, 8 * 32 * 256 * 4, stream>>>(qb, kb, vb, o_buf);

  // 5) out = o @ w_o^T  -> fp32 d_out
  k_gemm_bf16<<<dim3((unsigned)(D / 64), (unsigned)(M / 128)), 256, gemm_lds, stream>>>(
      o_buf, wob, d_out, (int)M, (int)D, (int)D, 1);
}
